// Decoder_40398462386805
// MI455X (gfx1250) — matrix operand order repaired and checked by reading, not compiled, not run on hardware
//
#include <hip/hip_runtime.h>
#include <hip/hip_bf16.h>
#include <math.h>

// ---------------- problem constants ----------------
#define B_   8
#define S_   1024
#define D_   768
#define H_   4
#define DH_  192          // head dim
#define FF_  1024
#define NL_  3
#define EPS_ 1e-5f

// ---------------- WMMA tile config ----------------
#define BM   128
#define BN   64
#define BK   32
#define LDSK 40           // BK + 8 halves padding (80B row -> no bank conflicts)

// softmax chunking
#define QCH  8
#define QPC  (S_ / QCH)   // 128 query rows per chunk

typedef _Float16 half_t;
typedef __attribute__((ext_vector_type(16))) _Float16 v16h;
typedef __attribute__((ext_vector_type(8)))  float    v8f;

// ---- A fragment (16x32 f16), contiguous K order (matches B):
//   lanes 0-15 : row M = lane,   VGPR j: K=2j,2j+1
//   lanes 16-31: row M = lane-16, same but K offset +16
__device__ __forceinline__ v16h load_frag_a(const half_t* base, int lane) {
  const int hl = lane & 15, hi = lane >> 4;
  const half_t* rp = base + hl * LDSK;
  union { v16h h; unsigned u[8]; } f;
#pragma unroll
  for (int j = 0; j < 8; ++j) {
    const int kk = 2 * j + hi * 16;
    f.u[j] = *(const unsigned*)(rp + kk);      // 4B-aligned -> ds_load_b32/b128
  }
  return f.h;
}

// ---- B fragment (32x16 f16), column-per-lane (C/D-style striping):
//   lanes 0-15 : col N = lane, K=0..15 ; lanes 16-31: col N = lane-16, K=16..31
//   Bt is stored [n][k] so each lane reads 16 contiguous halves.
__device__ __forceinline__ v16h load_frag_b(const half_t* base, int lane) {
  const int hl = lane & 15, hi = lane >> 4;
  const half_t* cp = base + hl * LDSK + hi * 16;
  union { v16h h; unsigned u[8]; } f;
#pragma unroll
  for (int j = 0; j < 8; ++j) f.u[j] = *(const unsigned*)(cp + 2 * j);
  return f.h;
}

// =====================================================================
// Generic GEMM: C[M,N] = A[M,K] @ W[K,N] + bias, optional ReLU.
// Double-buffered LDS pipeline: one barrier per K-step, global prefetch
// of step k+1 overlaps the WMMAs of step k.
// grid = (N/BN, M/BM), block = 256 (8 waves, 4x2, 32x32 out per wave)
// =====================================================================
template <bool RELU>
__global__ __launch_bounds__(256) void gemm_bias_kernel(
    const float* __restrict__ A, const float* __restrict__ W,
    const float* __restrict__ bias, float* __restrict__ C,
    int M, int N, int K)
{
  __shared__ half_t As[2][BM * LDSK];
  __shared__ half_t Bt[2][BN * LDSK];
  const int tid = threadIdx.x;
  const int lane = tid & 31, wave = tid >> 5;
  const int waveM = wave >> 1, waveN = wave & 1;
  const int mBlk = blockIdx.y * BM, nBlk = blockIdx.x * BN;

  const int ar = tid >> 3, ac = (tid & 7) * 4;    // A tile 128x32
  const int br = tid >> 4, bc = (tid & 15) * 4;   // W tile 32x64

  float4 ra[4], rb[2];
  auto load_tiles = [&](int k0) {
#pragma unroll
    for (int i = 0; i < 4; ++i)
      ra[i] = *(const float4*)(A + (size_t)(mBlk + ar + i * 32) * K + k0 + ac);
#pragma unroll
    for (int i = 0; i < 2; ++i)
      rb[i] = *(const float4*)(W + (size_t)(k0 + br + i * 16) * N + nBlk + bc);
  };
  auto store_tiles = [&](int buf) {
#pragma unroll
    for (int i = 0; i < 4; ++i) {
      half_t* d = As[buf] + (ar + i * 32) * LDSK + ac;
      d[0] = (half_t)ra[i].x; d[1] = (half_t)ra[i].y;
      d[2] = (half_t)ra[i].z; d[3] = (half_t)ra[i].w;
    }
#pragma unroll
    for (int i = 0; i < 2; ++i) {
      const int kr = br + i * 16;
      Bt[buf][(bc + 0) * LDSK + kr] = (half_t)rb[i].x;
      Bt[buf][(bc + 1) * LDSK + kr] = (half_t)rb[i].y;
      Bt[buf][(bc + 2) * LDSK + kr] = (half_t)rb[i].z;
      Bt[buf][(bc + 3) * LDSK + kr] = (half_t)rb[i].w;
    }
  };

  v8f acc[2][2] = {};
  load_tiles(0);
  int buf = 0;
  for (int k0 = 0; k0 < K; k0 += BK) {
    store_tiles(buf);
    __syncthreads();
    if (k0 + BK < K) load_tiles(k0 + BK);   // prefetch next step

    v16h af[2], bf[2];
    af[0] = load_frag_a(As[buf] + (waveM * 32 +  0) * LDSK, lane);
    af[1] = load_frag_a(As[buf] + (waveM * 32 + 16) * LDSK, lane);
    bf[0] = load_frag_b(Bt[buf] + (waveN * 32 +  0) * LDSK, lane);
    bf[1] = load_frag_b(Bt[buf] + (waveN * 32 + 16) * LDSK, lane);
#pragma unroll
    for (int fm = 0; fm < 2; ++fm)
#pragma unroll
      for (int fn = 0; fn < 2; ++fn)
        acc[fm][fn] = __builtin_amdgcn_wmma_f32_16x16x32_f16(
            false, af[fm], false, bf[fn], (short)0, acc[fm][fn], false, false);
    buf ^= 1;
  }

  // C/D layout: lane&15 -> N ; VGPR r -> M = r (+8 for upper half-wave)
  const int hl = lane & 15, hi = lane >> 4;
#pragma unroll
  for (int fm = 0; fm < 2; ++fm)
#pragma unroll
    for (int fn = 0; fn < 2; ++fn) {
      const int col = nBlk + waveN * 32 + fn * 16 + hl;
      const float bv = bias[col];
#pragma unroll
      for (int r = 0; r < 8; ++r) {
        const int row = mBlk + waveM * 32 + fm * 16 + hi * 8 + r;
        float v = acc[fm][fn][r] + bv;
        if (RELU) v = fmaxf(v, 0.f);
        C[(size_t)row * N + col] = v;
      }
    }
}

// =====================================================================
// Attention scores: Sc[bh,q,k] = (Q_bh[q,:] . K_bh[k,:]) / 20, masked.
// grid = (S/BN, S/BM, B*H); double-buffered as above. 6 K-steps (DH=192).
// =====================================================================
__global__ __launch_bounds__(256) void attn_scores_kernel(
    const float* __restrict__ Q, const float* __restrict__ Km,
    const int* __restrict__ tokens, float* __restrict__ Sc)
{
  __shared__ half_t As[2][BM * LDSK];
  __shared__ half_t Bt[2][BN * LDSK];
  const int tid = threadIdx.x;
  const int lane = tid & 31, wave = tid >> 5;
  const int waveM = wave >> 1, waveN = wave & 1;
  const int mBlk = blockIdx.y * BM, nBlk = blockIdx.x * BN;
  const int bh = blockIdx.z, b = bh / H_, h = bh - b * H_;
  const int hb = h * DH_;

  const int ar = tid >> 3, ac = (tid & 7) * 4;    // 128 x 32 (queries)
  const int br = tid >> 3, bc = (tid & 7) * 4;    // 64 x 32 (keys, A-like rows)

  float4 ra[4], rb[2];
  auto load_tiles = [&](int k0) {
#pragma unroll
    for (int i = 0; i < 4; ++i)
      ra[i] = *(const float4*)(Q + (size_t)(b * S_ + mBlk + ar + i * 32) * D_ + hb + k0 + ac);
#pragma unroll
    for (int i = 0; i < 2; ++i)
      rb[i] = *(const float4*)(Km + (size_t)(b * S_ + nBlk + br + i * 32) * D_ + hb + k0 + bc);
  };
  auto store_tiles = [&](int buf) {
#pragma unroll
    for (int i = 0; i < 4; ++i) {
      half_t* d = As[buf] + (ar + i * 32) * LDSK + ac;
      d[0] = (half_t)ra[i].x; d[1] = (half_t)ra[i].y;
      d[2] = (half_t)ra[i].z; d[3] = (half_t)ra[i].w;
    }
#pragma unroll
    for (int i = 0; i < 2; ++i) {
      half_t* d = Bt[buf] + (br + i * 32) * LDSK + bc;   // K rows are K^T cols
      d[0] = (half_t)rb[i].x; d[1] = (half_t)rb[i].y;
      d[2] = (half_t)rb[i].z; d[3] = (half_t)rb[i].w;
    }
  };

  v8f acc[2][2] = {};
  load_tiles(0);
  int buf = 0;
  for (int k0 = 0; k0 < DH_; k0 += BK) {
    store_tiles(buf);
    __syncthreads();
    if (k0 + BK < DH_) load_tiles(k0 + BK);

    v16h af[2], bf[2];
    af[0] = load_frag_a(As[buf] + (waveM * 32 +  0) * LDSK, lane);
    af[1] = load_frag_a(As[buf] + (waveM * 32 + 16) * LDSK, lane);
    bf[0] = load_frag_b(Bt[buf] + (waveN * 32 +  0) * LDSK, lane);
    bf[1] = load_frag_b(Bt[buf] + (waveN * 32 + 16) * LDSK, lane);
#pragma unroll
    for (int fm = 0; fm < 2; ++fm)
#pragma unroll
      for (int fn = 0; fn < 2; ++fn)
        acc[fm][fn] = __builtin_amdgcn_wmma_f32_16x16x32_f16(
            false, af[fm], false, bf[fn], (short)0, acc[fm][fn], false, false);
    buf ^= 1;
  }

  const int hl = lane & 15, hi = lane >> 4;
  float* ScBH = Sc + (size_t)bh * S_ * S_;
#pragma unroll
  for (int fm = 0; fm < 2; ++fm)
#pragma unroll
    for (int fn = 0; fn < 2; ++fn) {
      const int kc = nBlk + waveN * 32 + fn * 16 + hl;
#pragma unroll
      for (int r = 0; r < 8; ++r) {
        const int q = mBlk + waveM * 32 + fm * 16 + hi * 8 + r;
        const bool masked = (kc > q) || (tokens[b * S_ + q] == 0);
        const float w = acc[fm][fn][r] * (1.0f / 20.0f);
        ScBH[(size_t)q * S_ + kc] = masked ? -1e9f : w;
      }
    }
}

// =====================================================================
// Softmax over the QUERY axis (faithful to nn.Softmax(dim=2)), chunked:
//  pass 1: per (bh, q-chunk, k) online running (max, sum)   [1 read]
//  pass 2: combine QCH partials per column                  [tiny]
//  pass 3: normalize                                        [1 read + 1 write]
// =====================================================================
__global__ __launch_bounds__(256) void softmax_partial_kernel(
    const float* __restrict__ Sc, float* __restrict__ pm, float* __restrict__ ps)
{
  // grid (S/256, QCH, B*H)
  const int k = blockIdx.x * 256 + threadIdx.x;
  const int qc = blockIdx.y, bh = blockIdx.z;
  const float* base = Sc + (size_t)bh * S_ * S_ + (size_t)qc * QPC * S_ + k;
  float m = -3.4e38f, s = 0.f;
  for (int q = 0; q < QPC; ++q) {
    const float v = base[(size_t)q * S_];
    const float mn = fmaxf(m, v);
    s = s * __expf(m - mn) + __expf(v - mn);
    m = mn;
  }
  const size_t idx = ((size_t)bh * QCH + qc) * S_ + k;
  pm[idx] = m; ps[idx] = s;
}

__global__ __launch_bounds__(256) void softmax_combine_kernel(
    const float* __restrict__ pm, const float* __restrict__ ps,
    float* __restrict__ fm, float* __restrict__ fs)
{
  // grid (S/256, B*H)
  const int k = blockIdx.x * 256 + threadIdx.x;
  const int bh = blockIdx.y;
  float m = -3.4e38f;
#pragma unroll
  for (int c = 0; c < QCH; ++c)
    m = fmaxf(m, pm[((size_t)bh * QCH + c) * S_ + k]);
  float s = 0.f;
#pragma unroll
  for (int c = 0; c < QCH; ++c) {
    const size_t i = ((size_t)bh * QCH + c) * S_ + k;
    s += ps[i] * __expf(pm[i] - m);
  }
  fm[(size_t)bh * S_ + k] = m;
  fs[(size_t)bh * S_ + k] = 1.0f / s;
}

__global__ __launch_bounds__(256) void softmax_norm_kernel(
    float* __restrict__ Sc, const float* __restrict__ fm, const float* __restrict__ fs)
{
  // grid (S/256, QCH, B*H)
  const int k = blockIdx.x * 256 + threadIdx.x;
  const int qc = blockIdx.y, bh = blockIdx.z;
  const float m = fm[(size_t)bh * S_ + k];
  const float inv = fs[(size_t)bh * S_ + k];
  float* base = Sc + (size_t)bh * S_ * S_ + (size_t)qc * QPC * S_ + k;
  for (int q = 0; q < QPC; ++q) {
    float* p = base + (size_t)q * S_;
    *p = __expf(*p - m) * inv;
  }
}

// =====================================================================
// attn output: X1[b,q,hb+n] = sum_k A[bh,q,k] * V[b,k,hb+n]
// grid = (DH/BN=3, S/BM=8, B*H); double-buffered; 32 K-steps.
// =====================================================================
__global__ __launch_bounds__(256) void attn_v_kernel(
    const float* __restrict__ Sc, const float* __restrict__ V,
    float* __restrict__ X1)
{
  __shared__ half_t As[2][BM * LDSK];
  __shared__ half_t Bt[2][BN * LDSK];
  const int tid = threadIdx.x;
  const int lane = tid & 31, wave = tid >> 5;
  const int waveM = wave >> 1, waveN = wave & 1;
  const int mBlk = blockIdx.y * BM, nBlk = blockIdx.x * BN;
  const int bh = blockIdx.z, b = bh / H_, h = bh - b * H_;
  const int hb = h * DH_;
  const float* A = Sc + (size_t)bh * S_ * S_;

  const int ar = tid >> 3, ac = (tid & 7) * 4;    // weights tile 128x32
  const int br = tid >> 4, bc = (tid & 15) * 4;   // V tile 32x64 (transposed)

  float4 ra[4], rb[2];
  auto load_tiles = [&](int k0) {
#pragma unroll
    for (int i = 0; i < 4; ++i)
      ra[i] = *(const float4*)(A + (size_t)(mBlk + ar + i * 32) * S_ + k0 + ac);
#pragma unroll
    for (int i = 0; i < 2; ++i)
      rb[i] = *(const float4*)(V + (size_t)(b * S_ + k0 + br + i * 16) * D_ + hb + nBlk + bc);
  };
  auto store_tiles = [&](int buf) {
#pragma unroll
    for (int i = 0; i < 4; ++i) {
      half_t* d = As[buf] + (ar + i * 32) * LDSK + ac;
      d[0] = (half_t)ra[i].x; d[1] = (half_t)ra[i].y;
      d[2] = (half_t)ra[i].z; d[3] = (half_t)ra[i].w;
    }
#pragma unroll
    for (int i = 0; i < 2; ++i) {
      const int kr = br + i * 16;
      Bt[buf][(bc + 0) * LDSK + kr] = (half_t)rb[i].x;
      Bt[buf][(bc + 1) * LDSK + kr] = (half_t)rb[i].y;
      Bt[buf][(bc + 2) * LDSK + kr] = (half_t)rb[i].z;
      Bt[buf][(bc + 3) * LDSK + kr] = (half_t)rb[i].w;
    }
  };

  v8f acc[2][2] = {};
  load_tiles(0);
  int buf = 0;
  for (int k0 = 0; k0 < S_; k0 += BK) {
    store_tiles(buf);
    __syncthreads();
    if (k0 + BK < S_) load_tiles(k0 + BK);

    v16h af[2], bf[2];
    af[0] = load_frag_a(As[buf] + (waveM * 32 +  0) * LDSK, lane);
    af[1] = load_frag_a(As[buf] + (waveM * 32 + 16) * LDSK, lane);
    bf[0] = load_frag_b(Bt[buf] + (waveN * 32 +  0) * LDSK, lane);
    bf[1] = load_frag_b(Bt[buf] + (waveN * 32 + 16) * LDSK, lane);
#pragma unroll
    for (int fm = 0; fm < 2; ++fm)
#pragma unroll
      for (int fn = 0; fn < 2; ++fn)
        acc[fm][fn] = __builtin_amdgcn_wmma_f32_16x16x32_f16(
            false, af[fm], false, bf[fn], (short)0, acc[fm][fn], false, false);
    buf ^= 1;
  }

  const int hl = lane & 15, hi = lane >> 4;
#pragma unroll
  for (int fm = 0; fm < 2; ++fm)
#pragma unroll
    for (int fn = 0; fn < 2; ++fn) {
      const int col = hb + nBlk + waveN * 32 + fn * 16 + hl;
#pragma unroll
      for (int r = 0; r < 8; ++r) {
        const int q = mBlk + waveM * 32 + fm * 16 + hi * 8 + r;
        X1[(size_t)(b * S_ + q) * D_ + col] = acc[fm][fn][r];
      }
    }
}

// =====================================================================
// out = LayerNorm(X + Dl) * g + beta      (one block per row, D=768)
// Safe for Out == X (all reads captured before the first barrier).
// =====================================================================
__global__ __launch_bounds__(256) void add_ln_kernel(
    const float* __restrict__ X, const float* __restrict__ Dl,
    const float* __restrict__ g, const float* __restrict__ beta,
    float* __restrict__ Out)
{
  __shared__ float red1[256], red2[256];
  const size_t row = blockIdx.x;
  const float* xr = X + row * D_;
  const float* dr = Dl + row * D_;
  float v[3], s = 0.f, s2 = 0.f;
#pragma unroll
  for (int i = 0; i < 3; ++i) {
    const int c = threadIdx.x + i * 256;
    const float t = xr[c] + dr[c];
    v[i] = t; s += t; s2 += t * t;
  }
  red1[threadIdx.x] = s; red2[threadIdx.x] = s2;
  __syncthreads();
  for (int off = 128; off > 0; off >>= 1) {
    if (threadIdx.x < off) {
      red1[threadIdx.x] += red1[threadIdx.x + off];
      red2[threadIdx.x] += red2[threadIdx.x + off];
    }
    __syncthreads();
  }
  const float mean = red1[0] * (1.0f / D_);
  const float var  = red2[0] * (1.0f / D_) - mean * mean;
  const float inv  = rsqrtf(var + EPS_);
#pragma unroll
  for (int i = 0; i < 3; ++i) {
    const int c = threadIdx.x + i * 256;
    Out[row * D_ + c] = (v[i] - mean) * inv * g[c] + beta[c];
  }
}

// =====================================================================
// x[b,s,:] = tok_emb[tokens[b,s],:] + pos_emb[s,:]
// =====================================================================
__global__ __launch_bounds__(256) void embed_kernel(
    const int* __restrict__ tokens, const float* __restrict__ tok_emb,
    const float* __restrict__ pos_emb, float* __restrict__ X)
{
  const size_t row = blockIdx.x;               // b*S + s
  const int s = (int)(row % S_);
  const size_t tok = (size_t)tokens[row];
#pragma unroll
  for (int i = 0; i < 3; ++i) {
    const int c = threadIdx.x + i * 256;
    X[row * D_ + c] = tok_emb[tok * D_ + c] + pos_emb[(size_t)s * D_ + c];
  }
}

// =====================================================================
extern "C" void kernel_launch(void* const* d_in, const int* in_sizes, int n_in,
                              void* d_out, int out_size, void* d_ws, size_t ws_size,
                              hipStream_t stream) {
  (void)in_sizes; (void)n_in; (void)out_size; (void)ws_size;

  const int*   tokens  = (const int*)  d_in[0];
  const float* tok_emb = (const float*)d_in[1];
  const float* pos_emb = (const float*)d_in[2];
  const float* Wq = (const float*)d_in[3];
  const float* bq = (const float*)d_in[4];
  const float* Wk = (const float*)d_in[5];
  const float* bk = (const float*)d_in[6];
  const float* Wv = (const float*)d_in[7];
  const float* bv = (const float*)d_in[8];
  const float* ln_ag = (const float*)d_in[9];
  const float* ln_ab = (const float*)d_in[10];
  const float* W1 = (const float*)d_in[11];
  const float* b1 = (const float*)d_in[12];
  const float* W2 = (const float*)d_in[13];
  const float* b2 = (const float*)d_in[14];
  const float* ln_fg = (const float*)d_in[15];
  const float* ln_fb = (const float*)d_in[16];

  // workspace layout (floats)
  const size_t NX = (size_t)B_ * S_ * D_;          // 6,291,456
  float* ws = (float*)d_ws;
  float* x  = ws;
  float* qb = x  + NX;
  float* kb = qb + NX;
  float* vb = kb + NX;
  float* x1 = vb + NX;
  float* h1 = x1 + NX;                             // 8192 x 1024
  float* sc = h1 + (size_t)B_ * S_ * FF_;          // 32 x 1024 x 1024
  float* pm = sc + (size_t)B_ * H_ * S_ * S_;      // 32 x QCH x 1024
  float* ps = pm + (size_t)B_ * H_ * QCH * S_;
  float* fm = ps + (size_t)B_ * H_ * QCH * S_;     // 32 x 1024
  float* fs = fm + (size_t)B_ * H_ * S_;

  const int M = B_ * S_;                           // 8192
  const dim3 blk(256);
  const dim3 gProj(D_ / BN, M / BM);               // (12, 64)
  const dim3 gFF1(FF_ / BN, M / BM);               // (16, 64)
  const dim3 gFF2(D_ / BN, M / BM);                // (12, 64)
  const dim3 gSc(S_ / BN, S_ / BM, B_ * H_);       // (16, 8, 32)
  const dim3 gSmP(S_ / 256, QCH, B_ * H_);         // (4, 8, 32)
  const dim3 gSmC(S_ / 256, B_ * H_);              // (4, 32)
  const dim3 gAV(DH_ / BN, S_ / BM, B_ * H_);      // (3, 8, 32)

  embed_kernel<<<M, blk, 0, stream>>>(tokens, tok_emb, pos_emb, x);

  for (int l = 0; l < NL_; ++l) {
    for (int a = 0; a < 2; ++a) {
      const size_t wOff = ((size_t)l * 2 + a) * D_ * D_;
      const size_t bOff = ((size_t)l * 2 + a) * D_;
      gemm_bias_kernel<false><<<gProj, blk, 0, stream>>>(x, Wq + wOff, bq + bOff, qb, M, D_, D_);
      gemm_bias_kernel<false><<<gProj, blk, 0, stream>>>(x, Wk + wOff, bk + bOff, kb, M, D_, D_);
      gemm_bias_kernel<false><<<gProj, blk, 0, stream>>>(x, Wv + wOff, bv + bOff, vb, M, D_, D_);
      attn_scores_kernel<<<gSc, blk, 0, stream>>>(qb, kb, tokens, sc);
      softmax_partial_kernel<<<gSmP, blk, 0, stream>>>(sc, pm, ps);
      softmax_combine_kernel<<<gSmC, blk, 0, stream>>>(pm, ps, fm, fs);
      softmax_norm_kernel<<<gSmP, blk, 0, stream>>>(sc, fm, fs);
      attn_v_kernel<<<gAV, blk, 0, stream>>>(sc, vb, x1);
      add_ln_kernel<<<M, blk, 0, stream>>>(x, x1, ln_ag + bOff, ln_ab + bOff, x);
    }
    gemm_bias_kernel<true ><<<gFF1, blk, 0, stream>>>(x, W1 + (size_t)l * D_ * FF_, b1 + (size_t)l * FF_, h1, M, FF_, D_);
    gemm_bias_kernel<false><<<gFF2, blk, 0, stream>>>(h1, W2 + (size_t)l * FF_ * D_, b2 + (size_t)l * D_, x1, M, D_, FF_);
    float* dst = (l == NL_ - 1) ? (float*)d_out : x;
    add_ln_kernel<<<M, blk, 0, stream>>>(x, x1, ln_fg + (size_t)l * D_, ln_fb + (size_t)l * D_, dst);
  }
}
